// NISFlow_21638045237437
// MI455X (gfx1250) — matrix operand order repaired and checked by reading, not compiled, not run on hardware
//
#include <hip/hip_runtime.h>

typedef _Float16 v16h __attribute__((ext_vector_type(16)));
typedef _Float16 v8h  __attribute__((ext_vector_type(8)));
typedef float    v8f  __attribute__((ext_vector_type(8)));

// may-alias views for the per-wave union staging buffer (f16 h <-> f32 out)
typedef _Float16 f16a __attribute__((may_alias));
typedef float    f32a __attribute__((may_alias));
typedef v8h      v8ha __attribute__((may_alias));

#define NBINS   32
#define KS      72          // f16 activation/weight K-stride (halfs): 16B-aligned, conflict-free
#define OSTR    81          // f32 layer-4 output row stride (floats)
#define WBYTES  (32*OSTR*4) // 10368 B per-wave union buffer
#define WAVES_PER_BLOCK 4
#define THREADS (WAVES_PER_BLOCK*32)
#define ROWS_PER_BLOCK (WAVES_PER_BLOCK*32)

struct FlowParams {
  const float* W[4][4];
  const float* B[4][4];
};

// One MLP linear for this wave's 32 rows.
// A: f16 activations in wbuf, rows stride KS. B: transposed f16 weights sWt[col*KS+k].
// Fragment feeds are pure ds_load_b128:
//   A lane: halfs 0..7 = K kbase..kbase+7, halfs 8..15 = kbase+16..kbase+23 (two 16B runs)
//   B lane: halfs 0..7 = K kb..kb+7, halfs 8..15 = kb+16..kb+23 (two 16B runs)
template<int K, int O, int OPAD, bool RELU, bool OUTF32>
__device__ __forceinline__ void linear_wmma(
    const float* __restrict__ gW, const float* __restrict__ gB,
    _Float16* sWt, float* sBias, unsigned char* wbuf, int tid, int lane)
{
  f16a* hh = (f16a*)wbuf;
  f32a* of = (f32a*)wbuf;

  // Cooperative transposed weight stage (coalesced global reads, scattered LDS writes).
  for (int idx = tid; idx < K*O; idx += THREADS) {
    const int k = idx / O;
    const int o = idx - k*O;
    sWt[o*KS + k] = (_Float16)gW[idx];
  }
  if (OPAD > O) {
    for (int idx = tid; idx < (OPAD-O)*K; idx += THREADS) {
      const int o = O + idx / K;
      const int k = idx - (idx / K)*K;
      sWt[o*KS + k] = (_Float16)0.0f;
    }
  }
  if (tid < OPAD) sBias[tid] = (tid < O) ? gB[tid] : 0.0f;
  __syncthreads();

  constexpr int KG = K/32;
  constexpr int NB = OPAD/16;
  const int rl = lane & 15;
  const int hi = lane >> 4;   // 0: lanes 0-15, 1: lanes 16-31

  v16h af[2][KG];
  #pragma unroll
  for (int mt = 0; mt < 2; ++mt) {
    const int row = mt*16 + rl;
    #pragma unroll
    for (int kh = 0; kh < KG; ++kh) {
      const int kbase = kh*32 + hi*8;
      v8h a0 = *(const v8ha*)(hh + row*KS + kbase);
      v8h a1 = *(const v8ha*)(hh + row*KS + kbase + 16);
      af[mt][kh] = __builtin_shufflevector(a0, a1,
                     0,1,2,3,4,5,6,7,8,9,10,11,12,13,14,15);
    }
  }

  #pragma unroll
  for (int nb = 0; nb < NB; ++nb) {
    const int col = nb*16 + rl;
    v16h bf[KG];
    #pragma unroll
    for (int kh = 0; kh < KG; ++kh) {
      const int kb = kh*32 + hi*8;
      v8h b0 = *(const v8ha*)(sWt + col*KS + kb);
      v8h b1 = *(const v8ha*)(sWt + col*KS + kb + 16);
      bf[kh] = __builtin_shufflevector(b0, b1,
                 0,1,2,3,4,5,6,7,8,9,10,11,12,13,14,15);
    }
    const float bias = sBias[col];
    #pragma unroll
    for (int mt = 0; mt < 2; ++mt) {
      v8f c = {0.f,0.f,0.f,0.f,0.f,0.f,0.f,0.f};
      #pragma unroll
      for (int kh = 0; kh < KG; ++kh)
        c = __builtin_amdgcn_wmma_f32_16x16x32_f16(
                false, af[mt][kh], false, bf[kh], (short)0, c, false, false);
      // C/D layout: VGPR v holds M = mt*16 + 8*hi + v, N = col. Fuse bias (+ReLU).
      const int wrow = mt*16 + hi*8;
      #pragma unroll
      for (int v = 0; v < 8; ++v) {
        float val = c[v] + bias;
        if (RELU) val = fmaxf(val, 0.0f);
        if (OUTF32) of[(wrow + v)*OSTR + col] = val;           // final layer: f32 for spline
        else        hh[(wrow + v)*KS  + col] = (_Float16)val;  // hidden: f16 for next A-frags
      }
    }
  }
  __syncthreads();   // protect sWt before the next linear's weight stage
}

__global__ __launch_bounds__(THREADS)
void nisflow_kernel(const float* __restrict__ x, float* __restrict__ out,
                    FlowParams p)
{
  __shared__ __align__(16) _Float16 sWt[80*KS];                      // 11520 B
  __shared__ float sBias[80];
  __shared__ __align__(16) unsigned char sBuf[WAVES_PER_BLOCK*WBYTES]; // 41472 B

  const int tid  = threadIdx.x;
  const int lane = tid & 31;
  const int wave = tid >> 5;
  unsigned char* wbuf = sBuf + wave*WBYTES;

  const int row = blockIdx.x * ROWS_PER_BLOCK + wave*32 + lane;
  float z0 = x[row*2 + 0];
  float z1 = x[row*2 + 1];
  float logdet = 0.0f;

  for (int layer = 0; layer < 4; ++layer) {
    const float xa = (layer & 1) ? z1 : z0;
    const float xb = (layer & 1) ? z0 : z1;

    // ---- Gaussian bin encoder: centers linspace(0,1,32), sigma=1/32; f16 vector stores ----
    float esum = 0.0f;
    #pragma unroll
    for (int k = 0; k < NBINS; ++k) {
      const float d = (xa - (float)k*(1.0f/31.0f)) * 32.0f;
      esum += __expf(-0.5f*d*d);
    }
    const float einv = 1.0f/(esum + 1e-8f);
    f16a* hh = (f16a*)wbuf;
    #pragma unroll
    for (int c4 = 0; c4 < 4; ++c4) {
      v8h ev;
      #pragma unroll
      for (int i = 0; i < 8; ++i) {
        const int k = c4*8 + i;
        const float d = (xa - (float)k*(1.0f/31.0f)) * 32.0f;
        ev[i] = (_Float16)(__expf(-0.5f*d*d) * einv);
      }
      *(v8ha*)(hh + lane*KS + c4*8) = ev;
    }

    // ---- MLP 32->64->64->64->65 via WMMA ----
    linear_wmma<32,64,64,true ,false>(p.W[layer][0], p.B[layer][0], sWt, sBias, wbuf, tid, lane);
    linear_wmma<64,64,64,true ,false>(p.W[layer][1], p.B[layer][1], sWt, sBias, wbuf, tid, lane);
    linear_wmma<64,64,64,true ,false>(p.W[layer][2], p.B[layer][2], sWt, sBias, wbuf, tid, lane);
    linear_wmma<64,65,80,false,true >(p.W[layer][3], p.B[layer][3], sWt, sBias, wbuf, tid, lane);

    // ---- spline: one row per lane, o[0..31]=W_un, o[32..64]=V_un (f32) ----
    const f32a* o = (const f32a*)wbuf + lane*OSTR;
    float m = -1e30f;
    for (int k = 0; k < NBINS; ++k) m = fmaxf(m, o[k]);
    float s = 0.0f;
    for (int k = 0; k < NBINS; ++k) s += __expf(o[k]-m);
    const float invs = 1.0f/s;

    const float val = fminf(fmaxf(xb, 0.0f), 0.9999f);
    // edges = [0, cumsum(softmax)], edges[32] forced to 1.0
    float cum = 0.0f, edge_b = 0.0f;
    int cnt = 0;
    for (int k = 0; k <= NBINS; ++k) {
      const float e = (k == 0) ? 0.0f : ((k == NBINS) ? 1.0f : cum);
      if (val >= e) { cnt++; edge_b = e; }
      if (k < NBINS) cum += __expf(o[k]-m)*invs;
    }
    int bin = cnt - 1;
    bin = bin < 0 ? 0 : (bin > NBINS-1 ? NBINS-1 : bin);

    // trapezoid normalization + prefix area up to 'bin'
    float ev_prev = __expf(fminf(fmaxf(o[NBINS], -10.0f), 10.0f));
    float trap = 0.0f, pre = 0.0f;
    for (int k = 0; k < NBINS; ++k) {
      const float Wk = __expf(o[k]-m)*invs;
      const float ev_next = __expf(fminf(fmaxf(o[NBINS+k+1], -10.0f), 10.0f));
      const float t = 0.5f*(ev_prev + ev_next)*Wk;
      trap += t;
      if (k < bin) pre += t;
      ev_prev = ev_next;
    }
    const float invt = 1.0f/(trap + 1e-8f);
    const float w_b  = __expf(o[bin]-m)*invs;
    const float v_b  = __expf(fminf(fmaxf(o[NBINS+bin],   -10.0f), 10.0f))*invt;
    const float v_b1 = __expf(fminf(fmaxf(o[NBINS+bin+1], -10.0f), 10.0f))*invt;
    const float area_pre = pre*invt;
    const float alpha = (val - edge_b)/(w_b + 1e-8f);
    const float znew  = area_pre + alpha*v_b*w_b
                      + 0.5f*alpha*alpha*(v_b1 - v_b)*w_b;
    logdet += __logf(v_b + alpha*(v_b1 - v_b) + 1e-8f);

    if (layer & 1) z0 = znew; else z1 = znew;
    __syncthreads();   // spline reads done before next layer's encoder overwrites wbuf
  }
  out[row] = logdet;
}

extern "C" void kernel_launch(void* const* d_in, const int* in_sizes, int n_in,
                              void* d_out, int out_size, void* d_ws, size_t ws_size,
                              hipStream_t stream) {
  (void)n_in; (void)out_size; (void)d_ws; (void)ws_size;
  const float* x = (const float*)d_in[0];   // d_in[1] = condition: unused by reference
  FlowParams p;
  for (int i = 0; i < 4; ++i)
    for (int j = 0; j < 4; ++j) {
      p.W[i][j] = (const float*)d_in[2 + i*8 + j*2];
      p.B[i][j] = (const float*)d_in[2 + i*8 + j*2 + 1];
    }
  const int rows = in_sizes[0] / 2;          // N = 1,048,576
  const int grid = rows / ROWS_PER_BLOCK;    // exactly divisible (8192 blocks)
  nisflow_kernel<<<grid, THREADS, 0, stream>>>(x, (float*)d_out, p);
}
